// LSTM_72361609003239
// MI455X (gfx1250) — matrix operand order repaired and checked by reading, not compiled, not run on hardware
//
#include <hip/hip_runtime.h>
#include <hip/hip_bf16.h>
#include <math.h>

// LSTM: S=512, B=64, I=256, H=512. WMMA f16 (f32-accum) implementation for gfx1250.

#define S_LEN 512
#define BATCH 64
#define IN    256
#define HID   512
#define G4    (4 * HID)     // 2048 fused gate columns [f,i,o,g]
#define KTOT  (IN + HID)    // 768 fused K dimension [x_t | h]

typedef _Float16 h8v  __attribute__((ext_vector_type(8)));
typedef _Float16 h16v __attribute__((ext_vector_type(16)));
typedef float    f8v  __attribute__((ext_vector_type(8)));

// ---------------------------------------------------------------------------
// Build fused, transposed f16 weight matrix WT (G4 x KTOT):
//   WT[n][k] = k < IN ? Wx_gate(n)[k][col(n)] : Wh_gate(n)[k-IN][col(n)]
// ---------------------------------------------------------------------------
__global__ __launch_bounds__(256)
void build_wt_kernel(const float* __restrict__ Wxf, const float* __restrict__ Whf,
                     const float* __restrict__ Wxi, const float* __restrict__ Whi,
                     const float* __restrict__ Wxo, const float* __restrict__ Who,
                     const float* __restrict__ Wxg, const float* __restrict__ Whg,
                     _Float16* __restrict__ WT)
{
    size_t idx = (size_t)blockIdx.x * blockDim.x + threadIdx.x;
    if (idx >= (size_t)G4 * KTOT) return;
    int n = (int)(idx / KTOT);
    int k = (int)(idx % KTOT);
    int gate = n >> 9;          // 0..3 -> f,i,o,g
    int col  = n & (HID - 1);
    const float* Wx = (gate == 0) ? Wxf : (gate == 1) ? Wxi : (gate == 2) ? Wxo : Wxg;
    const float* Wh = (gate == 0) ? Whf : (gate == 1) ? Whi : (gate == 2) ? Who : Whg;
    float v = (k < IN) ? Wx[(size_t)k * HID + col]
                       : Wh[(size_t)(k - IN) * HID + col];
    WT[idx] = (_Float16)v;
}

// ---------------------------------------------------------------------------
// Convert x (S,B,I) f32 -> f16, vectorized by 4.
// ---------------------------------------------------------------------------
__global__ __launch_bounds__(256)
void convert_x_kernel(const float* __restrict__ x, _Float16* __restrict__ x16, size_t n4)
{
    size_t i = (size_t)blockIdx.x * blockDim.x + threadIdx.x;
    if (i >= n4) return;
    const float4 v = ((const float4*)x)[i];
    _Float16* o = x16 + i * 4;
    o[0] = (_Float16)v.x; o[1] = (_Float16)v.y;
    o[2] = (_Float16)v.z; o[3] = (_Float16)v.w;
}

// ---------------------------------------------------------------------------
// Fused bias bsum[n] = bx[n] + bh[n]; zero h ping-pong buffers and c state.
// ---------------------------------------------------------------------------
__global__ __launch_bounds__(256)
void init_state_kernel(const float* __restrict__ bxf, const float* __restrict__ bhf,
                       const float* __restrict__ bxi, const float* __restrict__ bhi,
                       const float* __restrict__ bxo, const float* __restrict__ bho,
                       const float* __restrict__ bxg, const float* __restrict__ bhg,
                       float* __restrict__ bsum,
                       _Float16* __restrict__ h0, _Float16* __restrict__ h1,
                       float* __restrict__ c)
{
    int i = blockIdx.x * blockDim.x + threadIdx.x;
    if (i < G4) {
        int gate = i >> 9;
        int col  = i & (HID - 1);
        float bx = (gate == 0) ? bxf[col] : (gate == 1) ? bxi[col] : (gate == 2) ? bxo[col] : bxg[col];
        float bh = (gate == 0) ? bhf[col] : (gate == 1) ? bhi[col] : (gate == 2) ? bho[col] : bhg[col];
        bsum[i] = bx + bh;
    }
    if (i < BATCH * HID) {
        h0[i] = (_Float16)0.0f;
        h1[i] = (_Float16)0.0f;
        c[i]  = 0.0f;
    }
}

// ---------------------------------------------------------------------------
// One LSTM timestep.
// Grid: (BATCH/16, HID/16). Block: 128 threads = 4 waves; wave g owns gate g.
// gates(64 x 2048) = [x_t | h] (64 x 768) @ WT^T, via v_wmma_f32_16x16x32_f16.
// ---------------------------------------------------------------------------
__global__ __launch_bounds__(128)
void lstm_step_kernel(const _Float16* __restrict__ x16,   // (S,B,IN)
                      const _Float16* __restrict__ WT,    // (G4,KTOT)
                      const float*    __restrict__ bsum,  // (G4)
                      const _Float16* __restrict__ hin,   // (B,HID)
                      _Float16*       __restrict__ hout,  // (B,HID)
                      float*          __restrict__ cbuf,  // (B,HID)
                      float*          __restrict__ hseq,  // (S,B,HID) -> d_out
                      int s)
{
    __shared__ float gl[4][16][16];   // gate pre-activations [gate][m][n]

    const int tid  = threadIdx.x;
    const int wave = tid >> 5;        // gate index 0..3
    const int lane = tid & 31;
    const int half = lane >> 4;       // 0 or 1
    const int l16  = lane & 15;
    const int m0   = blockIdx.x * 16; // batch row tile
    const int j0   = blockIdx.y * 16; // hidden col tile

    // A fragment source: row m = l16 for both lane halves (K range differs by half).
    const int am = m0 + l16;
    const _Float16* xrow = x16 + ((size_t)s * BATCH + am) * IN;
    const _Float16* hrow = hin + (size_t)am * HID;

    // B fragment source: lane's column n = l16; khalf = half.
    const int ncol = wave * HID + j0 + l16;
    const _Float16* wrow = WT + (size_t)ncol * KTOT;

    f8v acc = {};
    #pragma unroll
    for (int kk = 0; kk < KTOT; kk += 32) {
        // ---- A (16x32 f16): two 16B chunks, K = kk+16*half+[0..7] and +8 ----
        const _Float16* asrc = (kk < IN) ? (xrow + kk) : (hrow + (kk - IN));
        h8v alo = *(const h8v*)(asrc + 16 * half);
        h8v ahi = *(const h8v*)(asrc + 16 * half + 8);
        h16v a;
        #pragma unroll
        for (int e = 0; e < 8; ++e) { a[e] = alo[e]; a[8 + e] = ahi[e]; }

        // ---- B (32x16 f16): lane holds K = kk + 16*half + [0..15], col n ----
        h16v b = *(const h16v*)(wrow + kk + 16 * half);

        acc = __builtin_amdgcn_wmma_f32_16x16x32_f16(
                  /*neg_a=*/false, a, /*neg_b=*/false, b,
                  /*c_mod=*/(short)0, acc, /*reuse_a=*/false, /*reuse_b=*/false);
    }

    // C/D layout: vgpr r -> row r + 8*half, column l16. Add fused bias, stash in LDS.
    const float bias = bsum[ncol];
    #pragma unroll
    for (int r = 0; r < 8; ++r) {
        gl[wave][r + 8 * half][l16] = acc[r] + bias;
    }
    __syncthreads();

    // Elementwise gate combine: 256 tile elements, 128 threads x 2.
    #pragma unroll
    for (int it = 0; it < 2; ++it) {
        int idx = tid + it * 128;
        int m = idx >> 4;
        int n = idx & 15;
        float fpre = gl[0][m][n];
        float ipre = gl[1][m][n];
        float opre = gl[2][m][n];
        float gpre = gl[3][m][n];
        float fg = 1.0f / (1.0f + expf(-fpre));
        float ig = 1.0f / (1.0f + expf(-ipre));
        float og = 1.0f / (1.0f + expf(-opre));
        float gg = tanhf(gpre);

        int brow = m0 + m;
        int hcol = j0 + n;
        size_t cidx = (size_t)brow * HID + hcol;
        float cnew = fg * cbuf[cidx] + ig * gg;
        float hnew = og * tanhf(cnew);
        cbuf[cidx] = cnew;
        hout[cidx] = (_Float16)hnew;
        hseq[((size_t)s * BATCH + brow) * HID + hcol] = hnew;
    }
}

// ---------------------------------------------------------------------------
// Host-side launch
// ---------------------------------------------------------------------------
extern "C" void kernel_launch(void* const* d_in, const int* in_sizes, int n_in,
                              void* d_out, int out_size, void* d_ws, size_t ws_size,
                              hipStream_t stream)
{
    const float* x    = (const float*)d_in[0];
    const float* Wx_f = (const float*)d_in[1];
    const float* bx_f = (const float*)d_in[2];
    const float* Wh_f = (const float*)d_in[3];
    const float* bh_f = (const float*)d_in[4];
    const float* Wx_i = (const float*)d_in[5];
    const float* bx_i = (const float*)d_in[6];
    const float* Wh_i = (const float*)d_in[7];
    const float* bh_i = (const float*)d_in[8];
    const float* Wx_o = (const float*)d_in[9];
    const float* bx_o = (const float*)d_in[10];
    const float* Wh_o = (const float*)d_in[11];
    const float* bh_o = (const float*)d_in[12];
    const float* Wx_g = (const float*)d_in[13];
    const float* bx_g = (const float*)d_in[14];
    const float* Wh_g = (const float*)d_in[15];
    const float* bh_g = (const float*)d_in[16];

    float* out = (float*)d_out;

    // Workspace carve (all offsets 256B-aligned)
    char* ws = (char*)d_ws;
    const size_t WT_BYTES  = (size_t)G4 * KTOT * sizeof(_Float16);     //  3,145,728
    const size_t X16_BYTES = (size_t)S_LEN * BATCH * IN * sizeof(_Float16); // 16,777,216
    const size_t H_BYTES   = (size_t)BATCH * HID * sizeof(_Float16);   //     65,536
    const size_t C_BYTES   = (size_t)BATCH * HID * sizeof(float);      //    131,072

    _Float16* WT   = (_Float16*)(ws);
    _Float16* x16  = (_Float16*)(ws + WT_BYTES);
    _Float16* h0   = (_Float16*)(ws + WT_BYTES + X16_BYTES);
    _Float16* h1   = (_Float16*)(ws + WT_BYTES + X16_BYTES + H_BYTES);
    float*    cbuf = (float*)   (ws + WT_BYTES + X16_BYTES + 2 * H_BYTES);
    float*    bsum = (float*)   (ws + WT_BYTES + X16_BYTES + 2 * H_BYTES + C_BYTES);

    // 1) Fused transposed f16 weights
    {
        size_t n = (size_t)G4 * KTOT;
        int blocks = (int)((n + 255) / 256);
        build_wt_kernel<<<blocks, 256, 0, stream>>>(Wx_f, Wh_f, Wx_i, Wh_i,
                                                    Wx_o, Wh_o, Wx_g, Wh_g, WT);
    }
    // 2) x -> f16
    {
        size_t n4 = (size_t)S_LEN * BATCH * IN / 4;
        int blocks = (int)((n4 + 255) / 256);
        convert_x_kernel<<<blocks, 256, 0, stream>>>(x, x16, n4);
    }
    // 3) Fused bias + zero state
    {
        int n = BATCH * HID;  // 32768 >= G4
        init_state_kernel<<<(n + 255) / 256, 256, 0, stream>>>(
            bx_f, bh_f, bx_i, bh_i, bx_o, bh_o, bx_g, bh_g, bsum, h0, h1, cbuf);
    }
    // 4) Sequential recurrence: one kernel per timestep, h ping-pong in f16.
    {
        dim3 grid(BATCH / 16, HID / 16);
        for (int s = 0; s < S_LEN; ++s) {
            const _Float16* hin = (s & 1) ? h1 : h0;
            _Float16*      hout = (s & 1) ? h0 : h1;
            lstm_step_kernel<<<grid, 128, 0, stream>>>(x16, WT, bsum, hin, hout,
                                                       cbuf, out, s);
        }
    }
    // 5) Final h and c slots: h_final = h_seq[S-1]; c_final = cbuf.
    const size_t HSEQ = (size_t)S_LEN * BATCH * HID;
    const size_t BH   = (size_t)BATCH * HID;
    hipMemcpyAsync(out + HSEQ, out + HSEQ - BH, BH * sizeof(float),
                   hipMemcpyDeviceToDevice, stream);
    hipMemcpyAsync(out + HSEQ + BH, cbuf, BH * sizeof(float),
                   hipMemcpyDeviceToDevice, stream);
}
